// RNNModel_48412871360841
// MI455X (gfx1250) — matrix operand order repaired and checked by reading, not compiled, not run on hardware
//
#include <hip/hip_runtime.h>
#include <math.h>
#include <stdint.h>

// ---------------- problem constants ----------------
#define BB 256
#define TT 500
#define II 256
#define HH 256
#define OO 128
#define MTOT (BB * TT)          // 128000 rows, divisible by 16
#define MTILES (MTOT / 16)      // 8000 A panels
#define HSTRIDE 264             // padded LDS row stride (bf16) -> conflict-free b128 reads

typedef __bf16 bf16_t;
typedef __attribute__((ext_vector_type(16))) __bf16 v16bf;
typedef __attribute__((ext_vector_type(8)))  __bf16 v8bf;
typedef __attribute__((ext_vector_type(8)))  float  v8f;

#if __has_builtin(__builtin_amdgcn_tanhf)
__device__ __forceinline__ float fast_tanh(float x) { return __builtin_amdgcn_tanhf(x); }
#else
__device__ __forceinline__ float fast_tanh(float x) { return tanhf(x); }
#endif

// ---------------- CDNA5 async-to-LDS (inline asm; portable across toolchains) ---------
// LDS byte address = low 32 bits of generic shared pointer.
__device__ __forceinline__ void async_ld_b128(uint32_t lds_off, const void* gptr) {
    asm volatile("global_load_async_to_lds_b128 %0, %1, off"
                 :: "v"(lds_off), "v"((uint64_t)(uintptr_t)gptr)
                 : "memory");
}
__device__ __forceinline__ void wait_async0() {
    asm volatile("s_wait_asynccnt 0" ::: "memory");
}
__device__ __forceinline__ uint32_t lds_addr(const void* p) {
    return (uint32_t)(uintptr_t)p;
}

// ---------------- fragment loaders -----------------
// A (16x32, MxK): lane = M row; elems 0..7 hold K = 16*half + 0..7,
// elems 8..15 hold K = 16*half + 8..15  -> two contiguous 16B runs.
__device__ __forceinline__ v16bf load_a_frag_lds(const bf16_t* pan, int m, int half, int kbase) {
    const v8bf lo = *(const v8bf*)(pan + m * HSTRIDE + kbase + 16 * half);
    const v8bf hi = *(const v8bf*)(pan + m * HSTRIDE + kbase + 16 * half + 8);
    v16bf r;
#pragma unroll
    for (int e = 0; e < 8; ++e) { r[e] = lo[e]; r[8 + e] = hi[e]; }
    return r;
}

// B (32x16, KxN): lane = N column; element e holds K = 16*half + e
// -> one contiguous 32B run per lane:  W[n, kbase + 16*half .. +15]
__device__ __forceinline__ v16bf load_b_frag_g(const bf16_t* __restrict__ W,
                                               int ld, int n, int half, int kbase) {
    return *(const v16bf*)(W + (long)n * ld + kbase + 16 * half);
}

#define WMMA_BF16(a, b, c) \
    __builtin_amdgcn_wmma_f32_16x16x32_bf16(false, (a), false, (b), (short)0, (c), false, false)

// ---------------- f32 -> bf16 conversion ----------------
__global__ void cvt_f32_bf16(const float* __restrict__ in, bf16_t* __restrict__ out, long n) {
    long i = (long)blockIdx.x * blockDim.x + threadIdx.x;
    if (i < n) out[i] = (bf16_t)in[i];
}

// ---------------- persistent GEMM: C = act(A[M,256] * W[N,256]^T + bias0 + bias1) ----
// 8 waves/block; wave owns NT 16x16 N-tiles (NT*8*16 = N columns covered).
// W fragments register-resident for the whole kernel; A panels double-buffered in LDS
// via global_load_async_to_lds_b128, prefetching panel i+1 while panel i is consumed.
template <int NT>
__global__ __launch_bounds__(256) void gemm_bf16_wmma(
    const bf16_t* __restrict__ A, const bf16_t* __restrict__ W,
    const float* __restrict__ bias0, const float* __restrict__ bias1,
    float* __restrict__ C, int N, int act) {
    const int K = 256;
    __shared__ bf16_t apan[2][16 * HSTRIDE];

    const int lane = threadIdx.x & 31;
    const int wave = threadIdx.x >> 5;
    const int half = lane >> 4;
    const int lm   = lane & 15;

    // Register-resident W fragments: NT N-tiles x 8 K-steps.
    v16bf wfrag[NT][8];
#pragma unroll
    for (int nt = 0; nt < NT; ++nt) {
        const int n = (wave * NT + nt) * 16 + lm;
#pragma unroll
        for (int kk = 0; kk < 8; ++kk)
            wfrag[nt][kk] = load_b_frag_g(W, K, n, half, kk * 32);
    }

    // Stage helper: copy one 16x256 bf16 A panel (512 x 16B chunks) into LDS buffer.
    auto stage = [&](int buf, long tm) {
#pragma unroll
        for (int i = 0; i < 2; ++i) {
            const int c   = (int)threadIdx.x + i * 256;
            const int row = c >> 5;           // 0..15
            const int col = c & 31;           // 16B chunk within row
            async_ld_b128(lds_addr(&apan[buf][row * HSTRIDE]) + col * 16,
                          (const char*)(A + (tm * 16 + row) * (long)K) + col * 16);
        }
    };

    long tm = blockIdx.x;
    if (tm >= MTILES) return;
    stage(0, tm);
    wait_async0();
    __syncthreads();

    int cur = 0;
    for (; tm < MTILES; tm += gridDim.x) {
        const long tnext = tm + gridDim.x;
        if (tnext < MTILES) stage(cur ^ 1, tnext);   // overlap with compute

        // Hoist all A fragments: 16 ds_load_b128 in flight, then the WMMA chain.
        v16bf a[8];
#pragma unroll
        for (int kk = 0; kk < 8; ++kk)
            a[kk] = load_a_frag_lds(apan[cur], lm, half, kk * 32);

        v8f acc[NT];
#pragma unroll
        for (int nt = 0; nt < NT; ++nt) acc[nt] = (v8f){};
#pragma unroll
        for (int kk = 0; kk < 8; ++kk) {
#pragma unroll
            for (int nt = 0; nt < NT; ++nt)
                acc[nt] = WMMA_BF16(a[kk], wfrag[nt][kk], acc[nt]);
        }

#pragma unroll
        for (int nt = 0; nt < NT; ++nt) {
            const int n = (wave * NT + nt) * 16 + lm;
            float bsum = bias0 ? bias0[n] : 0.0f;
            if (bias1) bsum += bias1[n];
#pragma unroll
            for (int r = 0; r < 8; ++r) {
                float v = acc[nt][r] + bsum;
                if (act == 1) v = 1.0f / (1.0f + __expf(-v));   // sigmoid
                const long row = tm * 16 + r + 8 * half;        // C: m = r + 8*half
                C[row * (long)N + n] = v;
            }
        }

        wait_async0();       // next panel resident in this wave
        __syncthreads();     // ... and in all waves; also fences panel reuse
        cur ^= 1;
    }
}

// ---------------- recurrent layer: h_t = tanh(xp_t + h_{t-1} * Whh^T) ----------------
// 16 blocks x 8 waves; wave owns columns [w*32, w*32+32) as two WMMA tiles.
// Whh register-resident; h in LDS (bf16, padded); xp_t double-buffered in LDS via
// async-to-LDS, prefetching step t+1 while step t computes.
__global__ __launch_bounds__(256) void rnn_recurrent(
    const float* __restrict__ xp,      // [B,T,H] f32, biases pre-folded
    const bf16_t* __restrict__ Whh,    // [H,H] bf16 row-major
    bf16_t* __restrict__ hout,         // [B,T,H] bf16
    int T) {
    __shared__ bf16_t hcur[16 * HSTRIDE];
    __shared__ float  xbuf[2][16 * 256];

    const int lane = threadIdx.x & 31;
    const int wave = threadIdx.x >> 5;
    const int half = lane >> 4;
    const int lm   = lane & 15;
    const int b0   = blockIdx.x * 16;

    v16bf wfrag[2][8];
#pragma unroll
    for (int nt = 0; nt < 2; ++nt) {
        const int n = (wave * 2 + nt) * 16 + lm;
#pragma unroll
        for (int kk = 0; kk < 8; ++kk)
            wfrag[nt][kk] = load_b_frag_g(Whh, HH, n, half, kk * 32);
    }

    // Stage xp tile for step t: 16 rows x 1KB = 1024 x 16B chunks, 4 per thread.
    auto stage_xp = [&](int buf, int t) {
#pragma unroll
        for (int i = 0; i < 4; ++i) {
            const int c   = (int)threadIdx.x + i * 256;
            const int row = c >> 6;           // 0..15
            const int col = c & 63;           // 16B chunk within row
            async_ld_b128(lds_addr(&xbuf[buf][row * 256]) + col * 16,
                          (const char*)(xp + ((long)(b0 + row) * T + t) * HH) + col * 16);
        }
    };

    for (int i = threadIdx.x; i < 16 * HSTRIDE; i += 256) hcur[i] = (bf16_t)0.0f;  // h0 = 0
    stage_xp(0, 0);
    wait_async0();
    __syncthreads();

    int cur = 0;
    for (int t = 0; t < T; ++t) {
        if (t + 1 < T) stage_xp(cur ^ 1, t + 1);     // overlap with compute

        // Hoist all A fragments off the serial critical path: issue all 16
        // ds_load_b128 first, then run the dependent WMMA chain.
        v16bf a[8];
#pragma unroll
        for (int kk = 0; kk < 8; ++kk)
            a[kk] = load_a_frag_lds(hcur, lm, half, kk * 32);

        v8f acc0 = {}, acc1 = {};
#pragma unroll
        for (int kk = 0; kk < 8; ++kk) {
            acc0 = WMMA_BF16(a[kk], wfrag[0][kk], acc0);
            acc1 = WMMA_BF16(a[kk], wfrag[1][kk], acc1);
        }
        __syncthreads();   // all reads of hcur done before overwrite

#pragma unroll
        for (int nt = 0; nt < 2; ++nt) {
            const int n = (wave * 2 + nt) * 16 + lm;
            const v8f acc = nt ? acc1 : acc0;
#pragma unroll
            for (int r = 0; r < 8; ++r) {
                const int  m   = r + 8 * half;
                float v = fast_tanh(acc[r] + xbuf[cur][m * 256 + n]);
                bf16_t hv = (bf16_t)v;
                hcur[m * HSTRIDE + n] = hv;
                hout[((long)(b0 + m) * T + t) * HH + n] = hv;
            }
        }

        wait_async0();       // xp tile for t+1 resident (this wave's chunks)
        __syncthreads();     // ... all waves' chunks; also publishes hcur
        cur ^= 1;
    }
}

// ---------------- host side ----------------
extern "C" void kernel_launch(void* const* d_in, const int* in_sizes, int n_in,
                              void* d_out, int out_size, void* d_ws, size_t ws_size,
                              hipStream_t stream) {
    (void)in_sizes; (void)n_in; (void)out_size; (void)ws_size;

    const float* x     = (const float*)d_in[0];
    const float* W_ih0 = (const float*)d_in[1];
    const float* W_hh0 = (const float*)d_in[2];
    const float* b_ih0 = (const float*)d_in[3];
    const float* b_hh0 = (const float*)d_in[4];
    const float* W_ih1 = (const float*)d_in[5];
    const float* W_hh1 = (const float*)d_in[6];
    const float* b_ih1 = (const float*)d_in[7];
    const float* b_hh1 = (const float*)d_in[8];
    const float* fc_W  = (const float*)d_in[9];
    const float* fc_b  = (const float*)d_in[10];
    float* out = (float*)d_out;

    const long nAct = (long)BB * TT * HH;      // 32,768,000 (== B*T*I here)
    char* ws = (char*)d_ws;
    bf16_t* x_bf = (bf16_t*)ws;                               ws += nAct * 2;
    float*  xp   = (float*)ws;                                ws += nAct * 4;
    bf16_t* h1   = (bf16_t*)ws;                               ws += nAct * 2;
    bf16_t* h2   = (bf16_t*)ws;                               ws += nAct * 2;
    bf16_t* wih0 = (bf16_t*)ws;                               ws += HH * II * 2;
    bf16_t* whh0 = (bf16_t*)ws;                               ws += HH * HH * 2;
    bf16_t* wih1 = (bf16_t*)ws;                               ws += HH * HH * 2;
    bf16_t* whh1 = (bf16_t*)ws;                               ws += HH * HH * 2;
    bf16_t* fcw  = (bf16_t*)ws;

    // 1) conversions to bf16
    cvt_f32_bf16<<<(int)((nAct + 255) / 256), 256, 0, stream>>>(x, x_bf, nAct);
    cvt_f32_bf16<<<(HH * II + 255) / 256, 256, 0, stream>>>(W_ih0, wih0, HH * II);
    cvt_f32_bf16<<<(HH * HH + 255) / 256, 256, 0, stream>>>(W_hh0, whh0, HH * HH);
    cvt_f32_bf16<<<(HH * HH + 255) / 256, 256, 0, stream>>>(W_ih1, wih1, HH * HH);
    cvt_f32_bf16<<<(HH * HH + 255) / 256, 256, 0, stream>>>(W_hh1, whh1, HH * HH);
    cvt_f32_bf16<<<(OO * HH + 255) / 256, 256, 0, stream>>>(fc_W, fcw, OO * HH);

    const int gemmBlocks = 2000;   // persistent: 8000 panels / 2000 blocks = 4 each

    // 2) layer 0: xp = x * W_ih0^T + (b_ih0 + b_hh0)
    gemm_bf16_wmma<2><<<gemmBlocks, 256, 0, stream>>>(x_bf, wih0, b_ih0, b_hh0, xp, HH, 0);
    // 3) layer-0 recurrence -> h1 (bf16)
    rnn_recurrent<<<BB / 16, 256, 0, stream>>>(xp, whh0, h1, TT);

    // 4) layer 1 projection: xp = h1 * W_ih1^T + (b_ih1 + b_hh1)
    gemm_bf16_wmma<2><<<gemmBlocks, 256, 0, stream>>>(h1, wih1, b_ih1, b_hh1, xp, HH, 0);
    // 5) layer-1 recurrence -> h2 (bf16)
    rnn_recurrent<<<BB / 16, 256, 0, stream>>>(xp, whh1, h2, TT);

    // 6) FC head + sigmoid -> d_out
    gemm_bf16_wmma<1><<<gemmBlocks, 256, 0, stream>>>(h2, fcw, fc_b, nullptr, out, OO, 1);
}
